// CoordModulatedDeformableC3D_53300544143735
// MI455X (gfx1250) — matrix operand order repaired and checked by reading, not compiled, not run on hardware
//
#include <hip/hip_runtime.h>
#include <hip/hip_fp16.h>

typedef _Float16 f16;
typedef __attribute__((ext_vector_type(16))) _Float16 v16h;
typedef __attribute__((ext_vector_type(8)))  float    v8f;
typedef __attribute__((ext_vector_type(4)))  unsigned int u32x4;
typedef __attribute__((ext_vector_type(8)))  int          i32x8;
typedef __attribute__((ext_vector_type(4)))  int          i32x4;

// ---------------- problem constants ----------------
constexpr int BATCH = 2, CIN = 64, COUT = 64, DD = 8, HH = 32, WW = 32;
constexpr int SP    = DD * HH * WW;      // 8192 spatial
constexpr int KT    = 27;                // 3x3x3 taps
constexpr int COFF  = 67;                // cin + 3 coord channels
constexpr int NOFF  = 108;               // offset-conv out channels
constexpr int NOFF_PAD = 112;            // 7 * 16

// ---------------- workspace layout (f16 elements, then f32 pred) ----------------
constexpr size_t XH_ELEMS   = (size_t)BATCH * CIN * SP;        // 1,048,576 halves
constexpr size_t WOFF_ELEMS = (size_t)KT * 3 * 7 * 512;        // 290,304 halves (21*512 per tap)
constexpr size_t WDCN_ELEMS = (size_t)KT * 2 * 4 * 512;        // 110,592 halves (8*512 per tap)
constexpr size_t WS_XH   = 0;
constexpr size_t WS_WOFF = WS_XH + XH_ELEMS;
constexpr size_t WS_WDCN = WS_WOFF + WOFF_ELEMS;
constexpr size_t HALF_TOTAL = WS_WDCN + WDCN_ELEMS;
constexpr size_t WS_PRED_BYTE = ((HALF_TOTAL * 2 + 255) / 256) * 256;

// ---------------- TDM availability ----------------
#if defined(__gfx1250__) && __has_builtin(__builtin_amdgcn_tensor_load_to_lds) && \
    __has_builtin(__builtin_amdgcn_s_wait_tensorcnt)
#define USE_TDM 1
#else
#define USE_TDM 0
#endif

#if USE_TDM
// Issue one TDM load of `elems` contiguous f16 (elems < 65536) from global to LDS.
// D# per CDNA5 ISA: 1-row tile, data_size=2B.
__device__ __forceinline__ void tdm_load_f16(const f16* gsrc, const f16* lds_dst, unsigned elems) {
  unsigned long long ga = (unsigned long long)(uintptr_t)gsrc;
  unsigned lds_off = (unsigned)(uintptr_t)lds_dst;   // flat LDS addr: low 32 bits = byte offset
  u32x4 g0;
  g0[0] = 1u;                                         // count=1 (valid user descriptor)
  g0[1] = lds_off;                                    // lds_addr (bytes)
  g0[2] = (unsigned)ga;                               // global_addr[31:0]
  g0[3] = (unsigned)((ga >> 32) & 0x01FFFFFFu) | (2u << 30);  // addr[56:32] | type=2
  i32x8 g1;
  g1[0] = (int)(1u << 16);                            // workgroup_mask=0, data_size=1 (2B)
  g1[1] = (int)((elems & 0xFFFFu) << 16);             // tensor_dim0[15:0]  -> bits[63:48]
  g1[2] = (int)((elems >> 16) & 0xFFFFu) | (1 << 16); // tensor_dim0[31:16] | tensor_dim1=1
  g1[3] = (int)((elems & 0xFFFFu) << 16);             // tile_dim0 = elems  -> bits[127:112]
  g1[4] = 1;                                          // tile_dim1 = 1
  g1[5] = (int)elems;                                 // tensor_dim0_stride[31:0]
  g1[6] = 0;
  g1[7] = 0;
  i32x4 z4 = {0, 0, 0, 0};
#if __clang_major__ >= 23
  i32x8 z8 = {0, 0, 0, 0, 0, 0, 0, 0};
  __builtin_amdgcn_tensor_load_to_lds(g0, g1, z4, z4, z8, 0);
#else
  __builtin_amdgcn_tensor_load_to_lds(g0, g1, z4, z4, 0);
#endif
}
#endif

// Fallback / host-pass staging: cooperative vectorized copy (elems % 16 == 0).
__device__ __forceinline__ void coop_stage(f16* dst, const f16* src, int elems, int tid, int nthr) {
  for (int i = tid * 16; i < elems; i += nthr * 16)
    *(v16h*)(dst + i) = *(const v16h*)(src + i);
}

// ---------------- prep kernels ----------------
__global__ void k_x_to_half(const float* __restrict__ x, f16* __restrict__ xh, int n) {
  int i = blockIdx.x * blockDim.x + threadIdx.x;
  if (i < n) xh[i] = (f16)x[i];
}

// wp[((k*3+c)*7+t)*512 + lane*16 + e] = w_off[n=t*16+lane%16][ch=c*32+e+(lane/16)*16][tap k]
__global__ void k_pack_woff(const float* __restrict__ w, f16* __restrict__ wp) {
  int id = blockIdx.x * blockDim.x + threadIdx.x;
  if (id >= (int)WOFF_ELEMS) return;
  int e    = id & 15;
  int lane = (id >> 4) & 31;
  int rest = id >> 9;
  int t    = rest % 7;
  int kc   = rest / 7;
  int c    = kc % 3, k = kc / 3;
  int n    = t * 16 + (lane & 15);
  int ch   = c * 32 + e + ((lane >> 4) << 4);
  float v = 0.0f;
  if (n < NOFF && ch < COFF) v = w[((size_t)n * COFF + ch) * KT + k];
  wp[id] = (f16)v;
}

// wp[((k*2+cc)*4+t)*512 + lane*16 + e] = w_dcn[o=t*16+lane%16][ch=cc*32+e+(lane/16)*16][tap k]
__global__ void k_pack_wdcn(const float* __restrict__ w, f16* __restrict__ wp) {
  int id = blockIdx.x * blockDim.x + threadIdx.x;
  if (id >= (int)WDCN_ELEMS) return;
  int e    = id & 15;
  int lane = (id >> 4) & 31;
  int rest = id >> 9;
  int t    = rest % 4;
  int kc   = rest / 4;
  int cc   = kc % 2, k = kc / 2;
  int o    = t * 16 + (lane & 15);
  int ch   = cc * 32 + e + ((lane >> 4) << 4);
  wp[id] = (f16)w[((size_t)o * CIN + ch) * KT + k];
}

// ---------------- kernel A: offset-prediction conv (implicit GEMM, WMMA f16) ----------------
// Wave: 16 voxels (M) x 112 channels (7 N-tiles). Per-tap weights TDM-staged in LDS.
__global__ __launch_bounds__(256) void k_offset_conv(
    const f16* __restrict__ xh, const f16* __restrict__ wp,
    const float* __restrict__ b_off, float* __restrict__ pred)
{
  __shared__ __align__(32) f16 wsm[21 * 512];   // 3 chunks * 7 tiles * 512 = 21,504 B

  const int lane  = threadIdx.x & 31;
  const int wid   = threadIdx.x >> 5;
  const int gw    = blockIdx.x * 8 + wid;
  const int mbase = gw * 16;
  const int laneHalf = lane >> 4;
  const int vL = mbase + (lane & 15);      // A-matrix row for this lane
  const int b  = vL >> 13;
  const int s  = vL & (SP - 1);
  const int d  = s >> 10, h = (s >> 5) & 31, w = s & 31;

  v8f acc[7] = {};
  const size_t xbase_b = (size_t)b * CIN * SP;

  for (int k = 0; k < KT; ++k) {
    // ---- stage this tap's 21 B-tiles into LDS ----
    __syncthreads();                              // previous tap's reads done
#if USE_TDM
    if (wid == 0) {
      tdm_load_f16(wp + (size_t)k * 21 * 512, wsm, 21 * 512);
      __builtin_amdgcn_s_wait_tensorcnt(0);
    }
#else
    coop_stage(wsm, wp + (size_t)k * 21 * 512, 21 * 512, threadIdx.x, 256);
#endif
    __syncthreads();                              // weights visible

    const int kd = k / 9 - 1, kh = (k / 3) % 3 - 1, kw = k % 3 - 1;
    const int nd = d + kd, nh = h + kh, nw = w + kw;
    const bool valid = (nd >= 0) & (nd < DD) & (nh >= 0) & (nh < HH) & (nw >= 0) & (nw < WW);
    const int nidx = (nd * HH + nh) * WW + nw;
    // coord channels: linspace(-1,1) -> constant-reciprocal FMA (no divides)
    const float cz = fmaf((float)nd, 2.0f / (float)(DD - 1), -1.0f);
    const float cy = fmaf((float)nh, 2.0f / (float)(HH - 1), -1.0f);
    const float cx = fmaf((float)nw, 2.0f / (float)(WW - 1), -1.0f);
    const f16* __restrict__ xp = xh + xbase_b + nidx;

    for (int c = 0; c < 3; ++c) {
      // Build A fragment: 16x32 f16, k = 16*laneHalf + e (contiguous order, same as packed B).
      v16h a;
      #pragma unroll
      for (int e = 0; e < 16; ++e) {
        const int ch = c * 32 + e + laneHalf * 16;
        float v = 0.0f;
        if (valid) {
          if (ch < CIN)       v = (float)xp[(size_t)ch * SP];
          else if (ch == 64)  v = cz;
          else if (ch == 65)  v = cy;
          else if (ch == 66)  v = cx;
        }
        a[e] = (f16)v;
      }
      // Preload all 7 B fragments from LDS, then burst the WMMAs.
      const f16* wpk = wsm + (size_t)(c * 7) * 512 + (size_t)lane * 16;
      v16h bf[7];
      #pragma unroll
      for (int t = 0; t < 7; ++t) bf[t] = *(const v16h*)(wpk + (size_t)t * 512);
      #pragma unroll
      for (int t = 0; t < 7; ++t)
        acc[t] = __builtin_amdgcn_wmma_f32_16x16x32_f16(
            false, a, false, bf[t], (short)0, acc[t], false, false);
    }
  }

  // Epilogue: bias; sigmoid fused for alpha channels (>=81); store [voxel][112] f32.
  #pragma unroll
  for (int t = 0; t < 7; ++t) {
    const int n = t * 16 + (lane & 15);
    if (n < NOFF) {
      const float bias = b_off[n];
      #pragma unroll
      for (int r = 0; r < 8; ++r) {
        const int mv = mbase + r + laneHalf * 8;
        float v = acc[t][r] + bias;
        if (n >= 81) v = 1.0f / (1.0f + __expf(-v));
        pred[(size_t)mv * NOFF_PAD + n] = v;
      }
    }
  }
}

// ---------------- kernel C: modulated deformable conv (gather + WMMA f16) ----------------
// Wave: 16 voxels (M) x 64 outputs (4 N-tiles). K = 27 taps * 64 ch. Per-tap weights in LDS.
__global__ __launch_bounds__(256) void k_deform_conv(
    const f16* __restrict__ xh, const f16* __restrict__ wp,
    const float* __restrict__ pred, const float* __restrict__ b_dcn,
    float* __restrict__ out)
{
  __shared__ __align__(32) f16 wsm[8 * 512];    // 2 chunks * 4 tiles * 512 = 8,192 B

  const int lane  = threadIdx.x & 31;
  const int wid   = threadIdx.x >> 5;
  const int gw    = blockIdx.x * 8 + wid;
  const int mbase = gw * 16;
  const int laneHalf = lane >> 4;
  const int vL = mbase + (lane & 15);
  const int b  = vL >> 13;
  const int s  = vL & (SP - 1);
  const float fd = (float)(s >> 10);
  const float fh = (float)((s >> 5) & 31);
  const float fw = (float)(s & 31);

  const float* __restrict__ prow = pred + (size_t)vL * NOFF_PAD;
  __builtin_prefetch(prow, 0, 1);               // global_prefetch: pred row toward WGP

  v8f acc[4] = {};
  const size_t xb = (size_t)b * CIN * SP;

  for (int k = 0; k < KT; ++k) {
    // ---- stage this tap's 8 B-tiles into LDS ----
    __syncthreads();
#if USE_TDM
    if (wid == 0) {
      tdm_load_f16(wp + (size_t)k * 8 * 512, wsm, 8 * 512);
      __builtin_amdgcn_s_wait_tensorcnt(0);
    }
#else
    coop_stage(wsm, wp + (size_t)k * 8 * 512, 8 * 512, threadIdx.x, 256);
#endif
    __syncthreads();

    const float offz = prow[k * 3 + 0];
    const float offy = prow[k * 3 + 1];
    const float offx = prow[k * 3 + 2];
    const float alpha = prow[81 + k];

    const float pz = fd + (float)(k / 9 - 1) + offz;
    const float py = fh + (float)((k / 3) % 3 - 1) + offy;
    const float px = fw + (float)(k % 3 - 1) + offx;
    const float fz = floorf(pz), fy = floorf(py), fx = floorf(px);

    int   sidx[8];
    float wc[8];
    #pragma unroll
    for (int ci = 0; ci < 8; ++ci) {
      const float iz = fz + (float)(ci >> 2);
      const float iy = fy + (float)((ci >> 1) & 1);
      const float ix = fx + (float)(ci & 1);
      const float wgt = (1.0f - fabsf(pz - iz)) * (1.0f - fabsf(py - iy)) * (1.0f - fabsf(px - ix));
      const bool v = (iz >= 0.0f) & (iz < (float)DD) &
                     (iy >= 0.0f) & (iy < (float)HH) &
                     (ix >= 0.0f) & (ix < (float)WW);
      const int izc = (int)fminf(fmaxf(iz, 0.0f), (float)(DD - 1));
      const int iyc = (int)fminf(fmaxf(iy, 0.0f), (float)(HH - 1));
      const int ixc = (int)fminf(fmaxf(ix, 0.0f), (float)(WW - 1));
      sidx[ci] = (izc * HH + iyc) * WW + ixc;
      wc[ci]   = v ? wgt * alpha : 0.0f;        // alpha modulation folded into corner weights
    }

    for (int cc = 0; cc < 2; ++cc) {
      v16h a;
      #pragma unroll
      for (int e = 0; e < 16; ++e) {
        const int ch = cc * 32 + e + laneHalf * 16;
        const f16* __restrict__ xc = xh + xb + (size_t)ch * SP;
        float sacc = 0.0f;
        #pragma unroll
        for (int ci = 0; ci < 8; ++ci)
          sacc += wc[ci] * (float)xc[sidx[ci]];
        a[e] = (f16)sacc;
      }
      const f16* wpk = wsm + (size_t)(cc * 4) * 512 + (size_t)lane * 16;
      v16h bf[4];
      #pragma unroll
      for (int t = 0; t < 4; ++t) bf[t] = *(const v16h*)(wpk + (size_t)t * 512);
      #pragma unroll
      for (int t = 0; t < 4; ++t)
        acc[t] = __builtin_amdgcn_wmma_f32_16x16x32_f16(
            false, a, false, bf[t], (short)0, acc[t], false, false);
    }
  }

  // Epilogue: bias + store NCDHW fp32.
  #pragma unroll
  for (int t = 0; t < 4; ++t) {
    const int o = t * 16 + (lane & 15);
    const float bias = b_dcn[o];
    #pragma unroll
    for (int r = 0; r < 8; ++r) {
      const int mv = mbase + r + laneHalf * 8;
      const int bb = mv >> 13, ss = mv & (SP - 1);
      out[((size_t)bb * COUT + o) * SP + ss] = acc[t][r] + bias;
    }
  }
}

// ---------------- launcher ----------------
extern "C" void kernel_launch(void* const* d_in, const int* in_sizes, int n_in,
                              void* d_out, int out_size, void* d_ws, size_t ws_size,
                              hipStream_t stream) {
  const float* x     = (const float*)d_in[0];
  const float* w_off = (const float*)d_in[1];
  const float* b_off = (const float*)d_in[2];
  const float* w_dcn = (const float*)d_in[3];
  const float* b_dcn = (const float*)d_in[4];
  float* out = (float*)d_out;

  f16* wsH   = (f16*)d_ws;
  f16* xh    = wsH + WS_XH;
  f16* woffp = wsH + WS_WOFF;
  f16* wdcnp = wsH + WS_WDCN;
  float* pred = (float*)((char*)d_ws + WS_PRED_BYTE);

  k_x_to_half<<<(int)((XH_ELEMS + 255) / 256), 256, 0, stream>>>(x, xh, (int)XH_ELEMS);
  k_pack_woff<<<(int)((WOFF_ELEMS + 255) / 256), 256, 0, stream>>>(w_off, woffp);
  k_pack_wdcn<<<(int)((WDCN_ELEMS + 255) / 256), 256, 0, stream>>>(w_dcn, wdcnp);

  // 16384 voxels / (16 per wave * 8 waves per block) = 128 blocks
  k_offset_conv<<<128, 256, 0, stream>>>(xh, woffp, b_off, pred);
  k_deform_conv<<<128, 256, 0, stream>>>(xh, wdcnp, pred, b_dcn, out);
}
